// DPOTNet_49495203119188
// MI455X (gfx1250) — matrix operand order repaired and checked by reading, not compiled, not run on hardware
//
#include <hip/hip_runtime.h>
#include <hip/hip_bf16.h>
#include <cstdint>

typedef __bf16 bf16;
typedef __attribute__((ext_vector_type(16))) __bf16 v16bf;
typedef __attribute__((ext_vector_type(8)))  float  v8f;

#define TWO_PI 6.2831853071795864f

// ---------- helpers ----------
__device__ __forceinline__ float gelu_f(float x) {
    float x3 = x * x * x;
    return 0.5f * x * (1.0f + tanhf(0.7978845608028654f * (x + 0.044715f * x3)));
}
__device__ __forceinline__ bf16 f2bf(float f) {
    unsigned u = __float_as_uint(f);
    unsigned r = u + 0x7fffu + ((u >> 16) & 1u);
    unsigned short h = (unsigned short)(r >> 16);
    return __builtin_bit_cast(__bf16, h);
}
__device__ __forceinline__ float bf2f(bf16 b) {
    unsigned short h = __builtin_bit_cast(unsigned short, b);
    return __uint_as_float(((unsigned)h) << 16);
}
__device__ __forceinline__ int imin(int a, int b) { return a < b ? a : b; }

// =====================================================================
// Generic NT GEMM: C[M,N] = alpha * A[M,K](bf16,row-major,lda)
//                           @ Bt[N,K]^T(bf16,N-major,ldb)  (+ beta*Cin)
// epilogue: (+bias[n]) -> (gelu?) -> (+resid) -> outF / outB
// K must be a multiple of 32; callers guarantee every launched M is a
// multiple of 128; ragged N uses row-clamped loads (garbage columns are
// masked at store time, so no EXEC predication needed -> WMMA-legal).
//
// CDNA5 data path: double-buffered LDS tiles filled with
// GLOBAL_LOAD_ASYNC_TO_LDS_B128 (ASYNCcnt), next tile issued before
// waiting on the current one (s_wait_asynccnt 2), then workgroup
// barrier, then ds_load fragments per ISA 7.12.2 layouts -> v_wmma.
// =====================================================================
union Frag32B { uint4 q[2]; v16bf v; };

__global__ void __launch_bounds__(256)
gemm_nt_bf16(int M, int N, int K,
             const bf16* __restrict__ A, int lda,
             const bf16* __restrict__ Bt, int ldb,
             const float* Cin, float alpha, int beta,
             const float* __restrict__ bias,
             const float* __restrict__ resid,
             float* outF, bf16* outB, int ldc, int act)
{
    __shared__ alignas(16) bf16 As[2][128][40];   // 32 K + 8 pad (16B); ping-pong
    __shared__ alignas(16) bf16 Bs[2][128][40];

    const int tid  = threadIdx.x;
    const int lane = tid & 31;
    const int wave = tid >> 5;
    const int waveM = wave & 1;   // 2 waves along M (64 rows each)
    const int waveN = wave >> 1;  // 4 waves along N (32 cols each)
    const int bm = blockIdx.x * 128;
    const int bn = blockIdx.y * 128;
    const int lr = lane & 15, half = lane >> 4;

    v8f acc[4][2];
#pragma unroll
    for (int i = 0; i < 4; i++)
#pragma unroll
        for (int j = 0; j < 2; j++)
#pragma unroll
            for (int r = 0; r < 8; r++) acc[i][j][r] = 0.0f;

    const int lrow = tid >> 1;          // 0..127
    const int lk   = (tid & 1) * 16;    // 0 or 16 bf16
    const int am   = imin(bm + lrow, M - 1);   // row-clamped, always in-bounds
    const int bnr  = imin(bn + lrow, N - 1);
    const bf16* gA = A  + (long)am  * lda + lk;
    const bf16* gB = Bt + (long)bnr * ldb + lk;

    // async copy of one 128x32 bf16 K-tile pair into LDS buffer `buf`
    auto issue_tile = [&](int buf, int k0) {
        const unsigned la = (unsigned)(unsigned long long)&As[buf][lrow][lk];
        const unsigned lb = (unsigned)(unsigned long long)&Bs[buf][lrow][lk];
        asm volatile("global_load_async_to_lds_b128 %0, %1, off"
                     :: "v"(la), "v"(gA + k0) : "memory");
        asm volatile("global_load_async_to_lds_b128 %0, %1, off"
                     :: "v"(lb), "v"(gB + k0) : "memory");
    };

    issue_tile(0, 0);
    int buf = 0;
    for (int k0 = 0; k0 < K; k0 += 32, buf ^= 1) {
        const bool more = (k0 + 32) < K;
        if (more) {
            issue_tile(buf ^ 1, k0 + 32);            // prefetch next tile
            asm volatile("s_wait_asynccnt 0x2" ::: "memory");  // current tile landed
        } else {
            asm volatile("s_wait_asynccnt 0x0" ::: "memory");
        }
        __syncthreads();    // all waves' async tile data visible

        v16bf af[4], bfr[2];
#pragma unroll
        for (int ms = 0; ms < 4; ms++) {    // A: row m=lr, K = half*8+{0..7} then 16+half*8+{0..7}
            Frag32B f;
            const bf16* p = &As[buf][waveM * 64 + ms * 16 + lr][0];
            f.q[0] = *(const uint4*)(p + half * 8);
            f.q[1] = *(const uint4*)(p + half * 8 + 16);
            af[ms] = f.v;
        }
#pragma unroll
        for (int ns = 0; ns < 2; ns++) {    // B: col n=lr, K = half*8+{0..7} then 16+half*8+{0..7}
            Frag32B f;
            const bf16* p = &Bs[buf][waveN * 32 + ns * 16 + lr][0];
            f.q[0] = *(const uint4*)(p + half * 8);
            f.q[1] = *(const uint4*)(p + half * 8 + 16);
            bfr[ns] = f.v;
        }
#pragma unroll
        for (int ms = 0; ms < 4; ms++)
#pragma unroll
            for (int ns = 0; ns < 2; ns++)
                acc[ms][ns] = __builtin_amdgcn_wmma_f32_16x16x32_bf16(
                    false, af[ms], false, bfr[ns], (short)0, acc[ms][ns], false, false);

        __syncthreads();    // compute done before next issue overwrites other buffer
    }

#pragma unroll
    for (int ms = 0; ms < 4; ms++) {
#pragma unroll
        for (int ns = 0; ns < 2; ns++) {
            const int n = bn + waveN * 32 + ns * 16 + lr;
            if (n >= N) continue;
#pragma unroll
            for (int r = 0; r < 8; r++) {
                const int m = bm + waveM * 64 + ms * 16 + half * 8 + r;  // M = r + 8*half
                if (m >= M) continue;
                const long idx = (long)m * ldc + n;
                float v = alpha * acc[ms][ns][r];
                if (beta)  v += Cin[idx];
                if (bias)  v += bias[n];
                if (act)   v = gelu_f(v);
                if (resid) v += resid[idx];
                if (outF)  outF[idx] = v;
                if (outB)  outB[idx] = f2bf(v);
            }
        }
    }
}

// =====================================================================
// Patch embed for one frame t:  x(8,256,256,10,4)+grid -> bf16 A[8192][768]
// A is pre-scaled by t_embed[t,e] = cos((t/9)*2^linspace(-10,10,768)[e])
// =====================================================================
__global__ void __launch_bounds__(256)
patch_embed_kernel(const float* __restrict__ x,
                   const float* __restrict__ pe_w, const float* __restrict__ pe_b,
                   const float* __restrict__ pe1_w, const float* __restrict__ pe1_b,
                   const float* __restrict__ pos, bf16* __restrict__ Aout, int t)
{
    __shared__ float patch[448];   // 64 px * 7 ch
    __shared__ float inter[35];
    const int blk = blockIdx.x;                 // b*1024 + ph*32 + pw
    const int b = blk >> 10, ph = (blk >> 5) & 31, pw = blk & 31;
    const int tid = threadIdx.x;

    if (tid < 448) {
        const int pix = tid / 7, ci = tid % 7;
        const int pp = pix >> 3, qq = pix & 7;
        const int hh = ph * 8 + pp, ww = pw * 8 + qq;
        float v;
        if (ci < 4)       v = x[(((long)(b * 256 + hh) * 256 + ww) * 10 + t) * 4 + ci];
        else if (ci == 4) v = hh * (1.0f / 255.0f);
        else if (ci == 5) v = ww * (1.0f / 255.0f);
        else              v = t  * (1.0f / 9.0f);
        patch[tid] = v;
    }
    __syncthreads();
    if (tid < 35) {
        float s = pe_b[tid];
        for (int i = 0; i < 448; i++) s += patch[i] * pe_w[i * 35 + tid];
        inter[tid] = gelu_f(s);
    }
    __syncthreads();
    const float tl = t * (1.0f / 9.0f);
    const int posb = (ph * 32 + pw) * 768;
    const long rowb = (long)blk * 768;
    for (int j = 0; j < 3; j++) {
        const int e = tid * 3 + j;
        float s = pe1_b[e];
        for (int m = 0; m < 35; m++) s += inter[m] * pe1_w[m * 768 + e];
        s += pos[posb + e];
        const float gamma = exp2f(-10.0f + 20.0f * (float)e * (1.0f / 767.0f));
        Aout[rowb + e] = f2bf(s * cosf(tl * gamma));
    }
}

// ===================== GroupNorm (8 groups of 96 ch, per batch) ======
__global__ void __launch_bounds__(256)
groupnorm_kernel(const float* __restrict__ in, const float* __restrict__ w,
                 const float* __restrict__ bgn, float* outF, bf16* outB)
{
    const int b = blockIdx.x, g = blockIdx.y, tid = threadIdx.x;
    const long base = (long)b * 1024 * 768 + g * 96;
    float sum = 0.f, sq = 0.f;
    for (int i = tid; i < 1024 * 96; i += 256) {
        const int r = i / 96, c = i - r * 96;
        const float v = in[base + (long)r * 768 + c];
        sum += v; sq += v * v;
    }
    __shared__ float sbuf[256], qbuf[256];
    sbuf[tid] = sum; qbuf[tid] = sq;
    __syncthreads();
    for (int s = 128; s > 0; s >>= 1) {
        if (tid < s) { sbuf[tid] += sbuf[tid + s]; qbuf[tid] += qbuf[tid + s]; }
        __syncthreads();
    }
    const float mu = sbuf[0] * (1.0f / 98304.0f);
    const float var = qbuf[0] * (1.0f / 98304.0f) - mu * mu;
    const float rs = rsqrtf(var + 1e-5f);
    for (int i = tid; i < 1024 * 96; i += 256) {
        const int r = i / 96, c = i - r * 96;
        const long idx = base + (long)r * 768 + c;
        const int ch = g * 96 + c;
        const float v = (in[idx] - mu) * rs * w[ch] + bgn[ch];
        if (outF) outF[idx] = v;
        if (outB) outB[idx] = f2bf(v);
    }
}

// ===================== separable 32-pt DFT kernels (ortho) ===========
__global__ void __launch_bounds__(256)
rfft_w_kernel(const float* __restrict__ z, float* __restrict__ Fr, float* __restrict__ Fi)
{   // grid.x = B*32 (b,h), grid.y = 3 (c chunks)
    const int bh = blockIdx.x;
    const int c = blockIdx.y * 256 + threadIdx.x;
    float xs[32];
    const long ib = (long)bh * 32 * 768 + c;
#pragma unroll
    for (int w = 0; w < 32; w++) xs[w] = z[ib + (long)w * 768];
    const long ob = (long)bh * 17 * 768 + c;
    for (int k = 0; k < 17; k++) {
        float sr = 0.f, si = 0.f;
#pragma unroll
        for (int w = 0; w < 32; w++) {
            float s, cn; __sincosf(TWO_PI * (1.0f / 32.0f) * (float)(w * k), &s, &cn);
            sr += xs[w] * cn; si -= xs[w] * s;
        }
        Fr[ob + (long)k * 768] = sr * (1.0f / 32.0f);   // full ortho fwd scale here
        Fi[ob + (long)k * 768] = si * (1.0f / 32.0f);
    }
}

__global__ void __launch_bounds__(256)
fft_h_kernel(const float* __restrict__ Fr, const float* __restrict__ Fi,
             bf16* __restrict__ Xr, bf16* __restrict__ Xi)
{   // grid.x = B*17 (b,kw), grid.y = 3
    const int bk = blockIdx.x, b = bk / 17, kw = bk - b * 17;
    const int c = blockIdx.y * 256 + threadIdx.x;
    float fr[32], fi[32];
    const long base = ((long)b * 544 + kw) * 768 + c;
#pragma unroll
    for (int h = 0; h < 32; h++) { const long idx = base + (long)h * 17 * 768; fr[h] = Fr[idx]; fi[h] = Fi[idx]; }
    for (int kh = 0; kh < 32; kh++) {
        float ar = 0.f, ai = 0.f;
#pragma unroll
        for (int h = 0; h < 32; h++) {   // e^{-i theta}
            float s, cn; __sincosf(TWO_PI * (1.0f / 32.0f) * (float)(h * kh), &s, &cn);
            ar += fr[h] * cn + fi[h] * s;
            ai += fi[h] * cn - fr[h] * s;
        }
        const long idx = base + (long)kh * 17 * 768;
        Xr[idx] = f2bf(ar); Xi[idx] = f2bf(ai);
    }
}

__global__ void __launch_bounds__(256)
ifft_h_kernel(const float* __restrict__ Or, const float* __restrict__ Oi,
              float* __restrict__ Gr, float* __restrict__ Gi)
{   // grid.x = B*17, grid.y = 3
    const int bk = blockIdx.x, b = bk / 17, kw = bk - b * 17;
    const int c = blockIdx.y * 256 + threadIdx.x;
    float fr[32], fi[32];
    const long base = ((long)b * 544 + kw) * 768 + c;
#pragma unroll
    for (int kh = 0; kh < 32; kh++) { const long idx = base + (long)kh * 17 * 768; fr[kh] = Or[idx]; fi[kh] = Oi[idx]; }
    for (int h = 0; h < 32; h++) {
        float ar = 0.f, ai = 0.f;
#pragma unroll
        for (int kh = 0; kh < 32; kh++) {  // e^{+i theta}
            float s, cn; __sincosf(TWO_PI * (1.0f / 32.0f) * (float)(h * kh), &s, &cn);
            ar += fr[kh] * cn - fi[kh] * s;
            ai += fi[kh] * cn + fr[kh] * s;
        }
        const long idx = base + (long)h * 17 * 768;
        Gr[idx] = ar * (1.0f / 32.0f);     // full ortho inv scale here
        Gi[idx] = ai * (1.0f / 32.0f);
    }
}

__global__ void __launch_bounds__(256)
irfft_w_res_kernel(const float* __restrict__ Gr, const float* __restrict__ Gi,
                   const float* __restrict__ zgn, const float* __restrict__ Hin,
                   float* __restrict__ Hout)
{   // grid.x = B*32 (b,h), grid.y = 3; out = irfft + zgn (afno skip) + Hin (block skip)
    const int bh = blockIdx.x;
    const int c = blockIdx.y * 256 + threadIdx.x;
    float gr[17], gi[17];
    const long ib = (long)bh * 17 * 768 + c;
#pragma unroll
    for (int k = 0; k < 17; k++) { gr[k] = Gr[ib + (long)k * 768]; gi[k] = Gi[ib + (long)k * 768]; }
    const long tb = (long)bh * 32 * 768 + c;
    for (int w = 0; w < 32; w++) {
        float v = gr[0] + ((w & 1) ? -gr[16] : gr[16]);
#pragma unroll
        for (int k = 1; k < 16; k++) {
            float s, cn; __sincosf(TWO_PI * (1.0f / 32.0f) * (float)(w * k), &s, &cn);
            v += 2.0f * (gr[k] * cn - gi[k] * s);
        }
        const long idx = tb + (long)w * 768;
        Hout[idx] = v + zgn[idx] + Hin[idx];
    }
}

// ===================== classification head ===========================
__global__ void __launch_bounds__(256)
cls_kernel(const float* __restrict__ H,
           const float* __restrict__ w1, const float* __restrict__ b1,
           const float* __restrict__ w2, const float* __restrict__ b2,
           const float* __restrict__ w3, const float* __restrict__ b3,
           float* __restrict__ out)
{
    __shared__ float pooled[768], h1[768], h2[768];
    const int b = blockIdx.x, tid = threadIdx.x;
    for (int j = 0; j < 3; j++) {
        const int c = tid * 3 + j; float s = 0.f;
        for (int r = 0; r < 1024; r++) s += H[((long)b * 1024 + r) * 768 + c];
        pooled[c] = s * (1.0f / 1024.0f);
    }
    __syncthreads();
    for (int j = 0; j < 3; j++) {
        const int o = tid * 3 + j; float s = b1[o];
        for (int i = 0; i < 768; i++) s += pooled[i] * w1[i * 768 + o];
        h1[o] = gelu_f(s);
    }
    __syncthreads();
    for (int j = 0; j < 3; j++) {
        const int o = tid * 3 + j; float s = b2[o];
        for (int i = 0; i < 768; i++) s += h1[i] * w2[i * 768 + o];
        h2[o] = gelu_f(s);
    }
    __syncthreads();
    if (tid < 12) {
        float s = b3[tid];
        for (int i = 0; i < 768; i++) s += h2[i] * w3[i * 12 + tid];
        out[b * 12 + tid] = s;
    }
}

// ===================== small utility kernels =========================
__global__ void __launch_bounds__(256)
cast_transpose_kernel(const float* __restrict__ in, bf16* __restrict__ out, int K, int N)
{   // batched via blockIdx.y ; out[n*K+k] = in[k*N+n]
    const long off = (long)blockIdx.y * K * N;
    const long total = (long)K * N;
    for (long i = (long)blockIdx.x * 256 + threadIdx.x; i < total; i += (long)gridDim.x * 256) {
        const long k = i / N, n = i - k * N;
        out[off + n * (long)K + k] = f2bf(in[off + i]);
    }
}
__global__ void __launch_bounds__(256)
cast_f32_bf16_kernel(const float* __restrict__ in, bf16* __restrict__ out, long n)
{
    for (long i = (long)blockIdx.x * 256 + threadIdx.x; i < n; i += (long)gridDim.x * 256)
        out[i] = f2bf(in[i]);
}
__global__ void __launch_bounds__(256)
expand_dec_bias_kernel(const float* __restrict__ in, float* __restrict__ out)
{
    const int i = blockIdx.x * 256 + threadIdx.x;
    if (i < 2048) out[i] = in[i >> 6];   // n = o*64 + p*8 + q -> bias[o]
}
__global__ void __launch_bounds__(256)
dec_reorder_kernel(const bf16* __restrict__ Dec, float* __restrict__ Y)
{   // Y[b][hh][ww][o] = Dec[token(b,h,w)][o*64+p*8+q]  (gelu already applied)
    const long i = (long)blockIdx.x * 256 + threadIdx.x;
    if (i >= 16777216L) return;
    const int o = (int)(i & 31);
    long r = i >> 5;
    const int ww = (int)(r & 255); r >>= 8;
    const int hh = (int)(r & 255); const int b = (int)(r >> 8);
    const int h = hh >> 3, p = hh & 7, w = ww >> 3, q = ww & 7;
    const long token = ((long)b * 32 + h) * 32 + w;
    const long n = (long)o * 64 + p * 8 + q;
    Y[i] = bf2f(Dec[token * 2048 + n]);
}

// ===================== 3x3 SAME conv (32 in ch) ======================
template<int CO>
__global__ void __launch_bounds__(256)
conv3x3_kernel(const float* __restrict__ in, const float* __restrict__ w,
               const float* __restrict__ bias, float* __restrict__ out, int doGelu)
{   // grid: (W/64, H, B); weights HWIO staged in LDS, 3-row input halo tile in LDS
    __shared__ float win[3][66][32];
    __shared__ float wsm[9 * 32 * CO];
    const int tid = threadIdx.x;
    const int b = blockIdx.z, hh = blockIdx.y, c0 = blockIdx.x * 64;
    for (int i = tid; i < 9 * 32 * CO; i += 256) wsm[i] = w[i];
    for (int i = tid; i < 3 * 66 * 32; i += 256) {
        const int ci = i & 31;
        const int col = (i >> 5) % 66;
        const int ry = i / (66 * 32);
        const int gy = hh + ry - 1, gx = c0 + col - 1;
        float v = 0.f;
        if (gy >= 0 && gy < 256 && gx >= 0 && gx < 256)
            v = in[(((long)b * 256 + gy) * 256 + gx) * 32 + ci];
        win[ry][col][ci] = v;
    }
    __syncthreads();
    const int col = tid & 63, grp = tid >> 6;
    const int COT = CO / 4;
    float acc[COT];
#pragma unroll
    for (int j = 0; j < COT; j++) acc[j] = bias[grp * COT + j];
    for (int ky = 0; ky < 3; ky++)
        for (int kx = 0; kx < 3; kx++)
#pragma unroll 4
            for (int ci = 0; ci < 32; ci++) {
                const float v = win[ky][col + kx][ci];
                const float* wp = &wsm[((ky * 3 + kx) * 32 + ci) * CO + grp * COT];
#pragma unroll
                for (int j = 0; j < COT; j++) acc[j] += v * wp[j];
            }
    const long ob = (((long)b * 256 + hh) * 256 + (c0 + col)) * CO + grp * COT;
    for (int j = 0; j < COT; j++) {
        float v = acc[j];
        if (doGelu) v = gelu_f(v);
        out[ob + j] = v;
    }
}

// =====================================================================
// Host orchestration
// =====================================================================
extern "C" void kernel_launch(void* const* d_in, const int* in_sizes, int n_in,
                              void* d_out, int out_size, void* d_ws, size_t ws_size,
                              hipStream_t stream)
{
    (void)in_sizes; (void)n_in; (void)out_size; (void)ws_size;
    const float* x        = (const float*)d_in[0];
    const float* pe_w     = (const float*)d_in[1];
    const float* pe_b     = (const float*)d_in[2];
    const float* pe1_w    = (const float*)d_in[3];
    const float* pe1_b    = (const float*)d_in[4];
    const float* pos_emb  = (const float*)d_in[5];
    const float* ta_w     = (const float*)d_in[6];
    const float* n1_w     = (const float*)d_in[7];
    const float* n1_b     = (const float*)d_in[8];
    const float* afno_w1  = (const float*)d_in[9];
    const float* afno_b1  = (const float*)d_in[10];
    const float* afno_w2  = (const float*)d_in[11];
    const float* afno_b2  = (const float*)d_in[12];
    const float* n2_w     = (const float*)d_in[13];
    const float* n2_b     = (const float*)d_in[14];
    const float* m1_w     = (const float*)d_in[15];
    const float* m1_b     = (const float*)d_in[16];
    const float* m2_w     = (const float*)d_in[17];
    const float* m2_b     = (const float*)d_in[18];
    const float* cls_w1   = (const float*)d_in[19];
    const float* cls_b1   = (const float*)d_in[20];
    const float* cls_w2   = (const float*)d_in[21];
    const float* cls_b2   = (const float*)d_in[22];
    const float* cls_w3   = (const float*)d_in[23];
    const float* cls_b3   = (const float*)d_in[24];
    const float* dec_w    = (const float*)d_in[25];
    const float* dec_b    = (const float*)d_in[26];
    const float* oc1_w    = (const float*)d_in[27];
    const float* oc1_b    = (const float*)d_in[28];
    const float* oc2_w    = (const float*)d_in[29];
    const float* oc2_b    = (const float*)d_in[30];
    float* dout = (float*)d_out;

    // ---- workspace bump allocator ----
    char* wp = (char*)d_ws;
    auto alloc = [&](size_t bytes) -> void* {
        void* r = wp; wp += (bytes + 255) & ~(size_t)255; return r;
    };
    bf16*  taWt   = (bf16*) alloc(10ll * 768 * 768 * 2);
    bf16*  m1t    = (bf16*) alloc(4ll  * 768 * 768 * 2);
    bf16*  m2t    = (bf16*) alloc(4ll  * 768 * 768 * 2);
    bf16*  w1t    = (bf16*) alloc(64ll * 96 * 96 * 2);
    bf16*  w2t    = (bf16*) alloc(64ll * 96 * 96 * 2);
    bf16*  decWt  = (bf16*) alloc(2048ll * 768 * 2);
    float* decBias= (float*)alloc(2048 * 4);
    bf16*  Abuf   = (bf16*) alloc(8192ll * 768 * 2);   // per-frame patch embed / H bf16
    float* Hlat   = (float*)alloc(8192ll * 768 * 4);
    float* Hmid   = (float*)alloc(8192ll * 768 * 4);
    float* Zgn    = (float*)alloc(8192ll * 768 * 4);
    bf16*  Z2b    = (bf16*) alloc(8192ll * 768 * 2);
    bf16*  Mtmp   = (bf16*) alloc(8192ll * 768 * 2);
    float* Fr     = (float*)alloc(4352ll * 768 * 4);   // also reused as Gr
    float* Fi     = (float*)alloc(4352ll * 768 * 4);   // also reused as Gi
    bf16*  Xr     = (bf16*) alloc(4352ll * 768 * 2);
    bf16*  Xi     = (bf16*) alloc(4352ll * 768 * 2);
    bf16*  O1r    = (bf16*) alloc(4352ll * 768 * 2);
    bf16*  O1i    = (bf16*) alloc(4352ll * 768 * 2);
    float* ACC    = (float*)alloc(4352ll * 768 * 4);
    float* ACC2   = (float*)alloc(4352ll * 768 * 4);
    float* O2r    = (float*)alloc(4352ll * 768 * 4);
    float* O2i    = (float*)alloc(4352ll * 768 * 4);
    bf16*  Dec    = (bf16*) alloc(8192ll * 2048 * 2);
    float* Yimg   = (float*)alloc(8ll * 256 * 256 * 32 * 4);
    float* Yimg2  = (float*)alloc(8ll * 256 * 256 * 32 * 4);

    auto gemm = [&](int M, int N, int K, const bf16* A, int lda, const bf16* Bt, int ldb,
                    const float* Cin, float alpha, int beta, const float* bias,
                    const float* resid, float* outF, bf16* outB, int ldc, int act) {
        dim3 g((M + 127) / 128, (N + 127) / 128);
        gemm_nt_bf16<<<g, 256, 0, stream>>>(M, N, K, A, lda, Bt, ldb, Cin, alpha, beta,
                                            bias, resid, outF, outB, ldc, act);
    };

    // ---- weight prep: cast + transpose to bf16 K-major ----
    cast_transpose_kernel<<<dim3(2304, 10), 256, 0, stream>>>(ta_w,    taWt,  768, 768);
    cast_transpose_kernel<<<dim3(2304, 4),  256, 0, stream>>>(m1_w,    m1t,   768, 768);
    cast_transpose_kernel<<<dim3(2304, 4),  256, 0, stream>>>(m2_w,    m2t,   768, 768);
    cast_transpose_kernel<<<dim3(36, 64),   256, 0, stream>>>(afno_w1, w1t,   96,  96);
    cast_transpose_kernel<<<dim3(36, 64),   256, 0, stream>>>(afno_w2, w2t,   96,  96);
    cast_transpose_kernel<<<dim3(6144, 1),  256, 0, stream>>>(dec_w,   decWt, 768, 2048);
    expand_dec_bias_kernel<<<8, 256, 0, stream>>>(dec_b, decBias);

    // ---- patch embed + exp-time aggregator: H = sum_t (h0_t * temb_t) @ ta_w[t] ----
    for (int t = 0; t < 10; t++) {
        patch_embed_kernel<<<8192, 256, 0, stream>>>(x, pe_w, pe_b, pe1_w, pe1_b,
                                                     pos_emb, Abuf, t);
        gemm(8192, 768, 768, Abuf, 768, taWt + (long)t * 768 * 768, 768,
             Hlat, 1.0f, (t > 0) ? 1 : 0, nullptr, nullptr, Hlat, nullptr, 768, 0);
    }

    // ---- AFNO transformer blocks ----
    for (int d = 0; d < 4; d++) {
        groupnorm_kernel<<<dim3(8, 8), 256, 0, stream>>>(Hlat, n1_w + d * 768, n1_b + d * 768,
                                                         Zgn, nullptr);
        rfft_w_kernel<<<dim3(256, 3), 256, 0, stream>>>(Zgn, Fr, Fi);
        fft_h_kernel<<<dim3(136, 3), 256, 0, stream>>>(Fr, Fi, Xr, Xi);
        for (int n = 0; n < 8; n++) {
            const bf16* W1r = w1t + ((long)(d * 2 + 0) * 8 + n) * 9216;
            const bf16* W1i = w1t + ((long)(d * 2 + 1) * 8 + n) * 9216;
            const bf16* W2r = w2t + ((long)(d * 2 + 0) * 8 + n) * 9216;
            const bf16* W2i = w2t + ((long)(d * 2 + 1) * 8 + n) * 9216;
            const float* B1r = afno_b1 + ((d * 2 + 0) * 8 + n) * 96;
            const float* B1i = afno_b1 + ((d * 2 + 1) * 8 + n) * 96;
            const float* B2r = afno_b2 + ((d * 2 + 0) * 8 + n) * 96;
            const float* B2i = afno_b2 + ((d * 2 + 1) * 8 + n) * 96;
            const int o = n * 96;
            // layer1: o1r = gelu(Xr@W1r - Xi@W1i + b1r), o1i = gelu(Xi@W1r + Xr@W1i + b1i)
            gemm(4352, 96, 96, Xr + o, 768, W1r, 96, nullptr,  1.f, 0, nullptr, nullptr, ACC + o,  nullptr,  768, 0);
            gemm(4352, 96, 96, Xi + o, 768, W1i, 96, ACC + o, -1.f, 1, B1r,     nullptr, nullptr,  O1r + o,  768, 1);
            gemm(4352, 96, 96, Xi + o, 768, W1r, 96, nullptr,  1.f, 0, nullptr, nullptr, ACC2 + o, nullptr,  768, 0);
            gemm(4352, 96, 96, Xr + o, 768, W1i, 96, ACC2 + o, 1.f, 1, B1i,     nullptr, nullptr,  O1i + o,  768, 1);
            // layer2: o2r = O1r@W2r - O1i@W2i + b2r, o2i = O1i@W2r + O1r@W2i + b2i
            gemm(4352, 96, 96, O1r + o, 768, W2r, 96, nullptr,  1.f, 0, nullptr, nullptr, ACC + o,  nullptr, 768, 0);
            gemm(4352, 96, 96, O1i + o, 768, W2i, 96, ACC + o, -1.f, 1, B2r,     nullptr, O2r + o,  nullptr, 768, 0);
            gemm(4352, 96, 96, O1i + o, 768, W2r, 96, nullptr,  1.f, 0, nullptr, nullptr, ACC2 + o, nullptr, 768, 0);
            gemm(4352, 96, 96, O1r + o, 768, W2i, 96, ACC2 + o, 1.f, 1, B2i,     nullptr, O2i + o,  nullptr, 768, 0);
        }
        ifft_h_kernel<<<dim3(136, 3), 256, 0, stream>>>(O2r, O2i, Fr, Fi);
        irfft_w_res_kernel<<<dim3(256, 3), 256, 0, stream>>>(Fr, Fi, Zgn, Hlat, Hmid);
        groupnorm_kernel<<<dim3(8, 8), 256, 0, stream>>>(Hmid, n2_w + d * 768, n2_b + d * 768,
                                                         nullptr, Z2b);
        gemm(8192, 768, 768, Z2b,  768, m1t + (long)d * 768 * 768, 768, nullptr, 1.f, 0,
             m1_b + d * 768, nullptr, nullptr, Mtmp, 768, 1);
        gemm(8192, 768, 768, Mtmp, 768, m2t + (long)d * 768 * 768, 768, nullptr, 1.f, 0,
             m2_b + d * 768, Hmid, Hlat, nullptr, 768, 0);
    }

    // ---- classification head ----
    cls_kernel<<<8, 256, 0, stream>>>(Hlat, cls_w1, cls_b1, cls_w2, cls_b2, cls_w3, cls_b3,
                                      dout + 2097152);

    // ---- decoder: GEMM (8192 x 2048 x 768) + reorder + 3x3 convs ----
    cast_f32_bf16_kernel<<<4096, 256, 0, stream>>>(Hlat, Abuf, 8192L * 768);
    gemm(8192, 2048, 768, Abuf, 768, decWt, 768, nullptr, 1.f, 0, decBias, nullptr,
         nullptr, Dec, 2048, 1);
    dec_reorder_kernel<<<65536, 256, 0, stream>>>(Dec, Yimg);
    conv3x3_kernel<32><<<dim3(4, 256, 8), 256, 0, stream>>>(Yimg,  oc1_w, oc1_b, Yimg2, 1);
    conv3x3_kernel<4> <<<dim3(4, 256, 8), 256, 0, stream>>>(Yimg2, oc2_w, oc2_b, dout,  0);
}
